// HiFP8Uint8Linear_27539330302172
// MI455X (gfx1250) — matrix operand order repaired and checked by reading, not compiled, not run on hardware
//
#include <hip/hip_runtime.h>

// Quantized linear  y = x @ (u8 * scale[:,None]).T + bias
// GEMM: M=8192 (B*S), N=16384 (OUT), K=4096 (IN), compute-bound (AI ~1200).
// f16 WMMA, f32 accumulation; u8 exact in f16; scale/bias in epilogue.
// Block tile 256x128, 8 waves (4x2), wave tile 64x64 -> 16 WMMA / k-step.
// Double-buffered LDS, one barrier per k-step, register-staged global loads.

typedef __attribute__((ext_vector_type(16))) _Float16 v16h;
typedef __attribute__((ext_vector_type(8)))  _Float16 v8h;
typedef __attribute__((ext_vector_type(8)))  float    v8f;
typedef __attribute__((ext_vector_type(4)))  float    v4f;
typedef __attribute__((ext_vector_type(4)))  int      v4i;

#define M_TOTAL 8192
#define N_TOTAL 16384
#define K_TOTAL 4096
#define BM 256
#define BN 128
#define BK 32
#define KSTEPS (K_TOTAL / BK)   // 128

__global__ __launch_bounds__(256, 1)
void u8linear_wmma_kernel(const float* __restrict__ x,
                          const int*   __restrict__ w8,
                          const float* __restrict__ scale,
                          const float* __restrict__ bias,
                          float* __restrict__ out)
{
    // Double-buffered, fragment-swizzled tiles (ISA 7.12.2 f16 layouts):
    //   sA[buf][mtile(16)][lane(32)][16 halves]   A 16x32 fragment order
    //   sB[buf][ntile(8)][lane(32)][16 halves]    B 32x16 fragment order
    __shared__ alignas(32) _Float16 sA[2][BM * BK];   // 2 x 16 KB
    __shared__ alignas(32) _Float16 sB[2][BN * BK];   // 2 x  8 KB

    const int tid  = threadIdx.x;
    const int lane = tid & 31;
    const int wid  = tid >> 5;   // 0..7
    const int wm   = wid >> 1;   // 0..3 -> 64 M-rows per wave
    const int wn   = wid & 1;    // 0..1 -> 64 N-cols per wave

    // grid.x = M tiles (fastest): 2MB weight stripe stays L2-hot,
    // whole x (128MB) resident in 192MB L2.
    const int block_m = blockIdx.x * BM;
    const int block_n = blockIdx.y * BN;

    // ---- A staging: one thread per row, all 32 k-elements ----
    const float* xrow = x + (size_t)(block_m + tid) * K_TOTAL;
    const int amt = tid >> 4, aml = tid & 15;
    const int aoffLo = amt * 512 + (aml     ) * 16;  // lane<16 : k0-7 | k16-23
    const int aoffHi = amt * 512 + (aml + 16) * 16;  // lane>=16: k8-15| k24-31

    // ---- B staging: two threads per row, 16 k-elements ----
    const int srow = tid >> 1, shalf = tid & 1;
    const int* wrow = w8 + (size_t)(block_n + srow) * K_TOTAL + shalf * 16;
    const int boff = (srow >> 4) * 512 + (srow & 15) * 16 + shalf * 8;

    v4f fA[8];
    v4i iB[4];

    auto stage_load = [&](int kt) {
        const v4f* xp = (const v4f*)(xrow + kt * BK);
        const v4i* wp = (const v4i*)(wrow + kt * BK);
#pragma unroll
        for (int i = 0; i < 8; ++i) fA[i] = xp[i];
#pragma unroll
        for (int i = 0; i < 4; ++i) iB[i] = __builtin_nontemporal_load(&wp[i]);
    };

    auto stage_store = [&](int buf) {
        v8h aLo0, aLo1, aHi0, aHi1, d0, d1;
#pragma unroll
        for (int i = 0; i < 4; ++i) {
            aLo0[i] = (_Float16)fA[0][i]; aLo0[4 + i] = (_Float16)fA[1][i]; // k0-7
            aHi0[i] = (_Float16)fA[2][i]; aHi0[4 + i] = (_Float16)fA[3][i]; // k8-15
            aLo1[i] = (_Float16)fA[4][i]; aLo1[4 + i] = (_Float16)fA[5][i]; // k16-23
            aHi1[i] = (_Float16)fA[6][i]; aHi1[4 + i] = (_Float16)fA[7][i]; // k24-31
            d0[i] = (_Float16)(short)iB[0][i]; d0[4 + i] = (_Float16)(short)iB[1][i];
            d1[i] = (_Float16)(short)iB[2][i]; d1[4 + i] = (_Float16)(short)iB[3][i];
        }
        *(v8h*)&sA[buf][aoffLo]     = aLo0;
        *(v8h*)&sA[buf][aoffLo + 8] = aLo1;
        *(v8h*)&sA[buf][aoffHi]     = aHi0;
        *(v8h*)&sA[buf][aoffHi + 8] = aHi1;
        *(v8h*)&sB[buf][boff]       = d0;
        *(v8h*)&sB[buf][boff + 256] = d1;
    };

    v8f acc[4][4];
    {
        v8f z = {};
#pragma unroll
        for (int mi = 0; mi < 4; ++mi)
#pragma unroll
            for (int ni = 0; ni < 4; ++ni) acc[mi][ni] = z;
    }

    // prologue: fill buffer 0
    stage_load(0);
    stage_store(0);
    __syncthreads();

    for (int kt = 0; kt < KSTEPS; ++kt) {
        const int  cur  = kt & 1;
        const bool more = (kt + 1) < KSTEPS;

        // issue next-stage global loads early; they retire behind 16 WMMAs
        if (more) stage_load(kt + 1);

        v16h afrag[4];
#pragma unroll
        for (int mi = 0; mi < 4; ++mi)
            afrag[mi] = *(const v16h*)&sA[cur][(wm * 4 + mi) * 512 + lane * 16];

#pragma unroll
        for (int ni = 0; ni < 4; ++ni) {
            v16h bfrag = *(const v16h*)&sB[cur][(wn * 4 + ni) * 512 + lane * 16];
#pragma unroll
            for (int mi = 0; mi < 4; ++mi)
                acc[mi][ni] = __builtin_amdgcn_wmma_f32_16x16x32_f16(
                    false, afrag[mi], false, bfrag,
                    (short)0, acc[mi][ni], false, false);
        }

        // convert + store into the other buffer; single barrier per k-step
        if (more) stage_store(cur ^ 1);
        __syncthreads();
    }

    // epilogue: D layout (16x16 f32): M = r + 8*(lane>>4), N = lane&15
    const int l15 = lane & 15;
    const int lh8 = (lane >> 4) * 8;
#pragma unroll
    for (int ni = 0; ni < 4; ++ni) {
        const int col  = block_n + wn * 64 + ni * 16 + l15;
        const float sc = scale[col];
        const float bi = bias[col];
#pragma unroll
        for (int mi = 0; mi < 4; ++mi) {
            const int row0 = block_m + wm * 64 + mi * 16 + lh8;
#pragma unroll
            for (int r = 0; r < 8; ++r) {
                const float v = acc[mi][ni][r] * sc + bi;
                // non-temporal: 512MB output must not evict x from L2
                __builtin_nontemporal_store(v, &out[(size_t)(row0 + r) * N_TOTAL + col]);
            }
        }
    }
}

extern "C" void kernel_launch(void* const* d_in, const int* in_sizes, int n_in,
                              void* d_out, int out_size, void* d_ws, size_t ws_size,
                              hipStream_t stream) {
    (void)in_sizes; (void)n_in; (void)out_size; (void)d_ws; (void)ws_size;
    const float* x     = (const float*)d_in[0];
    const int*   w8    = (const int*)d_in[1];
    const float* scale = (const float*)d_in[2];
    const float* bias  = (const float*)d_in[3];
    float* out = (float*)d_out;

    dim3 grid(M_TOTAL / BM, N_TOTAL / BN);   // (32, 128), M fastest
    u8linear_wmma_kernel<<<grid, 256, 0, stream>>>(x, w8, scale, bias, out);
}
